// BinaryLinear_56453050138959
// MI455X (gfx1250) — matrix operand order repaired and checked by reading, not compiled, not run on hardware
//
#include <hip/hip_runtime.h>

// BinaryLinear: out[8192,4096] = x[8192,4096] @ sign(w)[4096,4096]^T + bias
// Compute-bound GEMM (275 GFLOP vs 14us memory floor) -> bf16 WMMA path.
// sign(w) in {-1,0,+1} is exact in bf16; x rounded RNE to bf16; fp32 accumulate.
//
// Preferred pipeline (if ws fits 100.7MB): convert x->bf16 and w->sign-bf16 once
// into d_ws, then run a bf16 GEMM whose K-loop uses GLOBAL_LOAD_ASYNC_TO_LDS
// (ASYNCcnt) so the inner loop is pure WMMA + ds_load fragments.
// Fallback: fused fp32->bf16 conversion inside the GEMM (no workspace needed).

#define M_DIM 8192
#define N_DIM 4096
#define K_DIM 4096

#define BM 128
#define BN 128

typedef __attribute__((ext_vector_type(16))) __bf16 bf16x16;
typedef __attribute__((ext_vector_type(8)))  float  f32x8;
typedef int v4i __attribute__((vector_size(16)));   // builtin's pointee type

union Frag16 {
    bf16x16 v;
    uint4   q[2];
};

struct __align__(8) h4 { __bf16 x, y, z, w; };
struct __align__(8) us4 { unsigned short x, y, z, w; };

__device__ __forceinline__ unsigned short sign_to_bf16_bits(float f) {
    unsigned int u = __float_as_uint(f);
    // sign(f): 0 -> 0.0, else +/-1.0bf16 (0x3F80 | signbit)
    return (u & 0x7FFFFFFFu) ? (unsigned short)(0x3F80u | ((u >> 16) & 0x8000u))
                             : (unsigned short)0;
}

// ---------------- async copy helpers (CDNA5 GLOBAL_LOAD_ASYNC_TO_LDS) --------

__device__ __forceinline__ void async_copy_b128(const void* g, void* l) {
#if __has_builtin(__builtin_amdgcn_global_load_async_to_lds_b128)
    __builtin_amdgcn_global_load_async_to_lds_b128(
        (__attribute__((address_space(1))) v4i*)g,
        (__attribute__((address_space(3))) v4i*)l, 0, 0);
#else
    *(uint4*)l = *(const uint4*)g;   // synchronous fallback
#endif
}

__device__ __forceinline__ void wait_async0() {
#if __has_builtin(__builtin_amdgcn_global_load_async_to_lds_b128)
#if __has_builtin(__builtin_amdgcn_s_wait_asynccnt)
    __builtin_amdgcn_s_wait_asynccnt(0);
#else
    asm volatile("s_wait_asynccnt 0" ::: "memory");
#endif
#endif
}

// ---------------- pass 1a: x fp32 -> bf16 (RNE via hardware cvt) -------------

__global__ __launch_bounds__(256)
void cvt_x_bf16(const float* __restrict__ in, __bf16* __restrict__ out) {
    const size_t i = ((size_t)blockIdx.x * 256 + threadIdx.x) * 8;
    float4 f0 = *(const float4*)(in + i);
    float4 f1 = *(const float4*)(in + i + 4);
    union { __bf16 h[8]; uint4 q; } u;
    u.h[0] = (__bf16)f0.x; u.h[1] = (__bf16)f0.y;
    u.h[2] = (__bf16)f0.z; u.h[3] = (__bf16)f0.w;
    u.h[4] = (__bf16)f1.x; u.h[5] = (__bf16)f1.y;
    u.h[6] = (__bf16)f1.z; u.h[7] = (__bf16)f1.w;
    *(uint4*)(out + i) = u.q;
}

// ---------------- pass 1b: w fp32 -> sign bf16 -------------------------------

__global__ __launch_bounds__(256)
void cvt_w_sign(const float* __restrict__ in, unsigned short* __restrict__ out) {
    const size_t i = ((size_t)blockIdx.x * 256 + threadIdx.x) * 8;
    float4 f0 = *(const float4*)(in + i);
    float4 f1 = *(const float4*)(in + i + 4);
    union { unsigned short h[8]; uint4 q; } u;
    u.h[0] = sign_to_bf16_bits(f0.x); u.h[1] = sign_to_bf16_bits(f0.y);
    u.h[2] = sign_to_bf16_bits(f0.z); u.h[3] = sign_to_bf16_bits(f0.w);
    u.h[4] = sign_to_bf16_bits(f1.x); u.h[5] = sign_to_bf16_bits(f1.y);
    u.h[6] = sign_to_bf16_bits(f1.z); u.h[7] = sign_to_bf16_bits(f1.w);
    *(uint4*)(out + i) = u.q;
}

// ---------------- pass 2: bf16 GEMM with async-LDS K-loop --------------------

#define BK2  64
#define LDT2 72   // BK2 + 8 pad elements -> 144B row stride (conflict-free b128)

__global__ __launch_bounds__(256, 2)
void gemm_bf16_async(const __bf16* __restrict__ Xb, const __bf16* __restrict__ Wb,
                     const float* __restrict__ Bias, float* __restrict__ Out)
{
    __shared__ __align__(16) __bf16 Als[2][BM * LDT2];
    __shared__ __align__(16) __bf16 Bls[2][BN * LDT2];

    const int tid   = threadIdx.x;
    const int lane  = tid & 31;
    const int wid   = tid >> 5;
    const int half  = lane >> 4;
    const int l16   = lane & 15;
    const int waveM = wid & 3;          // 4 waves x 32 rows
    const int waveN = wid >> 2;         // 2 waves x 64 cols

    const int blockN = blockIdx.x * BN;
    const int blockM = blockIdx.y * BM;

    f32x8 acc[2][4];
    #pragma unroll
    for (int i = 0; i < 2; ++i)
        #pragma unroll
        for (int j = 0; j < 4; ++j) {
            f32x8 z = {0.f, 0.f, 0.f, 0.f, 0.f, 0.f, 0.f, 0.f};
            acc[i][j] = z;
        }

    // Copy mapping: tile = 128 rows x 64 bf16 = 1024 x 16B chunks, 4 per thread.
    const __bf16* gA[4];
    const __bf16* gB[4];
    int lOff[4];
    #pragma unroll
    for (int i = 0; i < 4; ++i) {
        const int f   = tid + i * 256;
        const int row = f >> 3;         // 0..127
        const int c8  = (f & 7) * 8;    // element offset within row
        gA[i]   = Xb + (size_t)(blockM + row) * K_DIM + c8;
        gB[i]   = Wb + (size_t)(blockN + row) * K_DIM + c8;
        lOff[i] = row * LDT2 + c8;
    }

    // Issue k-tile 0 into buffer 0.
    #pragma unroll
    for (int i = 0; i < 4; ++i) {
        async_copy_b128(gA[i], &Als[0][lOff[i]]);
        async_copy_b128(gB[i], &Bls[0][lOff[i]]);
    }
    wait_async0();
    __syncthreads();

    const int NK = K_DIM / BK2;   // 64 k-steps
    int buf = 0;

    for (int kt = 0; kt < NK; ++kt) {
        // Issue next tile into the other buffer while WMMAs consume this one.
        if (kt + 1 < NK) {
            const int ke = (kt + 1) * BK2;
            #pragma unroll
            for (int i = 0; i < 4; ++i) {
                async_copy_b128(gA[i] + ke, &Als[buf ^ 1][lOff[i]]);
                async_copy_b128(gB[i] + ke, &Bls[buf ^ 1][lOff[i]]);
            }
        }

        // Two K=32 WMMA slices per k-step.
        #pragma unroll
        for (int ks = 0; ks < 2; ++ks) {
            // A (16x32 bf16): lane = row m; K = ks*32 + 8*half + {0..7}, then +16
            Frag16 afr[2];
            #pragma unroll
            for (int tm = 0; tm < 2; ++tm) {
                const __bf16* p = &Als[buf][(waveM * 32 + tm * 16 + l16) * LDT2 + ks * 32];
                afr[tm].q[0] = *(const uint4*)(p + 8 * half);
                afr[tm].q[1] = *(const uint4*)(p + 16 + 8 * half);
            }
            // B (32x16 bf16): lane = col n; K = ks*32 + 8*half + {0..7}, then +16
            Frag16 bfr[4];
            #pragma unroll
            for (int tn = 0; tn < 4; ++tn) {
                const __bf16* p = &Bls[buf][(waveN * 64 + tn * 16 + l16) * LDT2 + ks * 32];
                bfr[tn].q[0] = *(const uint4*)(p + 8 * half);
                bfr[tn].q[1] = *(const uint4*)(p + 16 + 8 * half);
            }
            #pragma unroll
            for (int tm = 0; tm < 2; ++tm)
                #pragma unroll
                for (int tn = 0; tn < 4; ++tn)
                    acc[tm][tn] = __builtin_amdgcn_wmma_f32_16x16x32_bf16(
                        false, afr[tm].v, false, bfr[tn].v,
                        (short)0, acc[tm][tn], false, false);
        }

        if (kt + 1 < NK) wait_async0();
        __syncthreads();
        buf ^= 1;
    }

    // Epilogue: C layout -> lane owns col l16, rows 8*half + r.
    #pragma unroll
    for (int tm = 0; tm < 2; ++tm) {
        #pragma unroll
        for (int tn = 0; tn < 4; ++tn) {
            const int gn  = blockN + waveN * 64 + tn * 16 + l16;
            const int gm0 = blockM + waveM * 32 + tm * 16 + half * 8;
            const float bv = Bias[gn];
            float* o = Out + (size_t)gm0 * N_DIM + gn;
            #pragma unroll
            for (int r = 0; r < 8; ++r)
                o[(size_t)r * N_DIM] = acc[tm][tn][r] + bv;
        }
    }
}

// ---------------- fallback: fused conversion GEMM (no workspace) -------------

#define BK1 32
#define LDT1 40

__global__ __launch_bounds__(256, 2)
void gemm_fused(const float* __restrict__ X, const float* __restrict__ W,
                const float* __restrict__ Bias, float* __restrict__ Out)
{
    __shared__ __align__(16) __bf16 Als[2][BM * LDT1];
    __shared__ __align__(16) __bf16 Bls[2][BN * LDT1];

    const int tid   = threadIdx.x;
    const int lane  = tid & 31;
    const int wid   = tid >> 5;
    const int half  = lane >> 4;
    const int l16   = lane & 15;
    const int waveM = wid & 3;
    const int waveN = wid >> 2;

    const int blockN = blockIdx.x * BN;
    const int blockM = blockIdx.y * BM;

    f32x8 acc[2][4];
    #pragma unroll
    for (int i = 0; i < 2; ++i)
        #pragma unroll
        for (int j = 0; j < 4; ++j) {
            f32x8 z = {0.f, 0.f, 0.f, 0.f, 0.f, 0.f, 0.f, 0.f};
            acc[i][j] = z;
        }

    int rowOf[4], colOf[4];
    #pragma unroll
    for (int i = 0; i < 4; ++i) {
        int f = tid + i * 256;
        rowOf[i] = f >> 3;
        colOf[i] = (f & 7) * 4;
    }

    float4 ax[4], bx[4];
    #pragma unroll
    for (int i = 0; i < 4; ++i) {
        ax[i] = *(const float4*)(X + (size_t)(blockM + rowOf[i]) * K_DIM + colOf[i]);
        bx[i] = *(const float4*)(W + (size_t)(blockN + rowOf[i]) * K_DIM + colOf[i]);
    }
    #pragma unroll
    for (int i = 0; i < 4; ++i) {
        h4 a4 = { (__bf16)ax[i].x, (__bf16)ax[i].y, (__bf16)ax[i].z, (__bf16)ax[i].w };
        us4 b4 = { sign_to_bf16_bits(bx[i].x), sign_to_bf16_bits(bx[i].y),
                   sign_to_bf16_bits(bx[i].z), sign_to_bf16_bits(bx[i].w) };
        *(h4*)&Als[0][rowOf[i] * LDT1 + colOf[i]]  = a4;
        *(us4*)&Bls[0][rowOf[i] * LDT1 + colOf[i]] = b4;
    }
    __syncthreads();

    const int NK = K_DIM / BK1;
    int buf = 0;

    for (int kt = 0; kt < NK; ++kt) {
        if (kt + 1 < NK) {
            const int k0 = (kt + 1) * BK1;
            #pragma unroll
            for (int i = 0; i < 4; ++i) {
                ax[i] = *(const float4*)(X + (size_t)(blockM + rowOf[i]) * K_DIM + k0 + colOf[i]);
                bx[i] = *(const float4*)(W + (size_t)(blockN + rowOf[i]) * K_DIM + k0 + colOf[i]);
            }
        }

        Frag16 afr[2];
        #pragma unroll
        for (int tm = 0; tm < 2; ++tm) {
            const __bf16* p = &Als[buf][(waveM * 32 + tm * 16 + l16) * LDT1];
            afr[tm].q[0] = *(const uint4*)(p + 8 * half);
            afr[tm].q[1] = *(const uint4*)(p + 16 + 8 * half);
        }
        Frag16 bfr[4];
        #pragma unroll
        for (int tn = 0; tn < 4; ++tn) {
            const __bf16* p = &Bls[buf][(waveN * 64 + tn * 16 + l16) * LDT1];
            bfr[tn].q[0] = *(const uint4*)(p + 8 * half);
            bfr[tn].q[1] = *(const uint4*)(p + 16 + 8 * half);
        }

        #pragma unroll
        for (int tm = 0; tm < 2; ++tm)
            #pragma unroll
            for (int tn = 0; tn < 4; ++tn)
                acc[tm][tn] = __builtin_amdgcn_wmma_f32_16x16x32_bf16(
                    false, afr[tm].v, false, bfr[tn].v,
                    (short)0, acc[tm][tn], false, false);

        if (kt + 1 < NK) {
            #pragma unroll
            for (int i = 0; i < 4; ++i) {
                h4 a4 = { (__bf16)ax[i].x, (__bf16)ax[i].y, (__bf16)ax[i].z, (__bf16)ax[i].w };
                us4 b4 = { sign_to_bf16_bits(bx[i].x), sign_to_bf16_bits(bx[i].y),
                           sign_to_bf16_bits(bx[i].z), sign_to_bf16_bits(bx[i].w) };
                *(h4*)&Als[buf ^ 1][rowOf[i] * LDT1 + colOf[i]]  = a4;
                *(us4*)&Bls[buf ^ 1][rowOf[i] * LDT1 + colOf[i]] = b4;
            }
        }
        __syncthreads();
        buf ^= 1;
    }

    #pragma unroll
    for (int tm = 0; tm < 2; ++tm) {
        #pragma unroll
        for (int tn = 0; tn < 4; ++tn) {
            const int gn  = blockN + waveN * 64 + tn * 16 + l16;
            const int gm0 = blockM + waveM * 32 + tm * 16 + half * 8;
            const float bv = Bias[gn];
            float* o = Out + (size_t)gm0 * N_DIM + gn;
            #pragma unroll
            for (int r = 0; r < 8; ++r)
                o[(size_t)r * N_DIM] = acc[tm][tn][r] + bv;
        }
    }
}

// ---------------- launch ------------------------------------------------------

extern "C" void kernel_launch(void* const* d_in, const int* in_sizes, int n_in,
                              void* d_out, int out_size, void* d_ws, size_t ws_size,
                              hipStream_t stream) {
    (void)in_sizes; (void)n_in; (void)out_size;
    const float* x    = (const float*)d_in[0];
    const float* w    = (const float*)d_in[1];
    const float* bias = (const float*)d_in[2];
    float*       out  = (float*)d_out;

    const size_t xElems = (size_t)M_DIM * K_DIM;            // 33.55M
    const size_t wElems = (size_t)N_DIM * K_DIM;            // 16.78M
    const size_t need   = (xElems + wElems) * sizeof(unsigned short);

    dim3 grid(N_DIM / BN, M_DIM / BM);   // 32 x 64 workgroups
    dim3 block(256);

    if (ws_size >= need) {
        __bf16* xb = (__bf16*)d_ws;
        __bf16* wb = (__bf16*)((char*)d_ws + xElems * sizeof(unsigned short));

        cvt_x_bf16<<<(unsigned)(xElems / (256 * 8)), 256, 0, stream>>>(x, xb);
        cvt_w_sign<<<(unsigned)(wElems / (256 * 8)), 256, 0, stream>>>(w, (unsigned short*)wb);
        gemm_bf16_async<<<grid, block, 0, stream>>>(xb, wb, bias, out);
    } else {
        gemm_fused<<<grid, block, 0, stream>>>(x, w, bias, out);
    }
}
